// GATv2Classifier_70720931496089
// MI455X (gfx1250) — matrix operand order repaired and checked by reading, not compiled, not run on hardware
//
#include <hip/hip_runtime.h>
#include <hip/hip_bf16.h>

typedef __attribute__((ext_vector_type(16))) _Float16 v16h;
typedef __attribute__((ext_vector_type(8)))  _Float16 v8h;
typedef __attribute__((ext_vector_type(8)))  float    v8f;
typedef __attribute__((ext_vector_type(4)))  float    v4f;

#define NNODES 50000
#define NEDGES 400000
#define NH 4
#define NC 64
#define HC 256   // NH*NC
#define NG 64
#define NEG_SLOPE 0.2f

// ---------------------------------------------------------------------------
// utility
// ---------------------------------------------------------------------------
__global__ void k_zero_f32(float* __restrict__ p, int n) {
  int i = blockIdx.x * blockDim.x + threadIdx.x;
  if (i < n) p[i] = 0.f;
}

// order-preserving float <-> uint mapping for atomicMax on floats
__device__ __forceinline__ unsigned f32_ord(float f) {
  unsigned u = __float_as_uint(f);
  return (u & 0x80000000u) ? ~u : (u | 0x80000000u);
}
__device__ __forceinline__ float ord_f32(unsigned e) {
  return (e & 0x80000000u) ? __uint_as_float(e & 0x7FFFFFFFu)
                           : __uint_as_float(~e);
}

// ---------------------------------------------------------------------------
// encoder: h = relu(x @ enc_w + enc_b), x:[N,8], enc_w:[8,64] -> f16 [N,64]
// ---------------------------------------------------------------------------
__global__ void k_encoder(const float* __restrict__ x, const float* __restrict__ w,
                          const float* __restrict__ b, _Float16* __restrict__ h) {
  int i = blockIdx.x * blockDim.x + threadIdx.x;
  if (i >= NNODES * 64) return;
  int c = i & 63, n = i >> 6;
  const float* xr = x + (size_t)n * 8;
  float acc = b[c];
#pragma unroll
  for (int k = 0; k < 8; ++k) acc += xr[k] * w[k * 64 + c];
  acc = fmaxf(acc, 0.f);
  h[(size_t)n * 64 + c] = (_Float16)acc;
}

// transpose+convert W f32 [K,HC] -> Wt f16 [HC,K] (contiguous K per output col)
__global__ void k_wconv(const float* __restrict__ w, _Float16* __restrict__ wt, int K) {
  int i = blockIdx.x * blockDim.x + threadIdx.x;
  if (i >= K * HC) return;
  int n = i % HC, k = i / HC;
  wt[(size_t)n * K + k] = (_Float16)w[(size_t)k * HC + n];
}

// ---------------------------------------------------------------------------
// WMMA GEMM: out[M,HC] = A[M,K](f16) * W[K,HC] + bias, via Wt[HC,K] f16.
// One wave computes a 64x64 supertile. Per 32-wide k-step: load 4 A-frags,
// then for each of 4 column sub-tiles load ONE B-frag and fire 4 WMMAs
// against it (only one B-frag live at a time -> no accumulator spills).
// __launch_bounds__(32,1): single wave, full 256-VGPR budget for the
// 128 accumulator VGPRs + frags.
// A-frag (16x32 f16, contiguous k order chosen here, same as B): lane L holds row M=L&15;
//   half i -> k = k0 + i + (L>=16)*16
// B-frag (32x16 f16): lane L holds col N=L&15; half i -> k = k0 + i + (L>=16)*16
// C/D f32: vgpr r, lane L -> row = r + (L>=16)*8, col = L&15
// ---------------------------------------------------------------------------
__global__ __launch_bounds__(32, 1)
void k_gemm_wmma(const _Float16* __restrict__ A,
                 const _Float16* __restrict__ Wt,
                 const float* __restrict__ bias,
                 float* __restrict__ out, int K) {
  int tileM = blockIdx.x;        // 64-row supertile
  int tileN = blockIdx.y;        // 64-col supertile
  int lane  = threadIdx.x;       // 0..31
  int hi    = lane >> 4;         // lane group
  int l16   = lane & 15;

  const _Float16* Abase = A  + (size_t)(tileM * 64 + l16) * K;
  const _Float16* Bbase = Wt + (size_t)(tileN * 64 + l16) * K;

  v8f c[4][4] = {};
#pragma unroll 2
  for (int k0 = 0; k0 < K; k0 += 32) {
    v16h a[4];
#pragma unroll
    for (int t = 0; t < 4; ++t) {
      const _Float16* Ap = Abase + (size_t)(t * 16) * K + k0;
      v8h lo  = *(const v8h*)(Ap + hi * 16);
      v8h hi8 = *(const v8h*)(Ap + hi * 16 + 8);
#pragma unroll
      for (int i = 0; i < 8; ++i) { a[t][i] = lo[i]; a[t][i + 8] = hi8[i]; }
    }
#pragma unroll
    for (int bn = 0; bn < 4; ++bn) {
      v16h b = *(const v16h*)(Bbase + (size_t)(bn * 16) * K + k0 + hi * 16);
#pragma unroll
      for (int am = 0; am < 4; ++am)
        c[am][bn] = __builtin_amdgcn_wmma_f32_16x16x32_f16(
            /*neg_a=*/false, a[am], /*neg_b=*/false, b,
            /*c_mod=*/(short)0, c[am][bn], /*reuse_a=*/false, /*reuse_b=*/false);
    }
  }

#pragma unroll
  for (int bn = 0; bn < 4; ++bn) {
    int n = tileN * 64 + bn * 16 + l16;
    float bv = bias ? bias[n] : 0.f;
#pragma unroll
    for (int am = 0; am < 4; ++am) {
      int mbase = tileM * 64 + am * 16 + hi * 8;
#pragma unroll
      for (int r = 0; r < 8; ++r) {
        int m = mbase + r;
        if (m < NNODES) out[(size_t)m * HC + n] = c[am][bn][r] + bv;
      }
    }
  }
}

// ---------------------------------------------------------------------------
// edge logits: per (edge, head)
// s = leaky_relu(xl[src] + xr[dst] + ea*we); logit = dot(s, att[h])
// ---------------------------------------------------------------------------
__global__ void k_edge_logits(const float* __restrict__ xl, const float* __restrict__ xr,
                              const float* __restrict__ ea, const float* __restrict__ we,
                              const float* __restrict__ att,
                              const int* __restrict__ src, const int* __restrict__ dst,
                              float* __restrict__ logits) {
  int i = blockIdx.x * blockDim.x + threadIdx.x;
  if (i >= NEDGES * NH) return;
  int h = i & 3, e = i >> 2;
  int s = src[e], d = dst[e];
  float eav = ea[e];
  const float* pl = xl + (size_t)s * HC + h * NC;
  const float* pr = xr + (size_t)d * HC + h * NC;
  const float* pw = we + h * NC;
  const float* pa = att + h * NC;
  float acc = 0.f;
#pragma unroll 4
  for (int c = 0; c < NC; ++c) {
    float t = pl[c] + pr[c] + eav * pw[c];
    t = (t > 0.f) ? t : NEG_SLOPE * t;
    acc += t * pa[c];
  }
  logits[(size_t)e * NH + h] = acc;
}

__global__ void k_segmax(const float* __restrict__ logits, const int* __restrict__ dst,
                         unsigned* __restrict__ mxu) {
  int i = blockIdx.x * blockDim.x + threadIdx.x;
  if (i >= NEDGES * NH) return;
  int h = i & 3, e = i >> 2;
  atomicMax(&mxu[(size_t)dst[e] * NH + h], f32_ord(logits[i]));
}

// exp(logit - max) in place, accumulate denom per (dst, head)
__global__ void k_expsum(float* __restrict__ logits, const int* __restrict__ dst,
                         const unsigned* __restrict__ mxu, float* __restrict__ denom) {
  int i = blockIdx.x * blockDim.x + threadIdx.x;
  if (i >= NEDGES * NH) return;
  int h = i & 3, e = i >> 2;
  int d = dst[e];
  float m  = ord_f32(mxu[(size_t)d * NH + h]);
  float ex = __expf(logits[i] - m);
  logits[i] = ex;
  atomicAdd(&denom[(size_t)d * NH + h], ex);
}

// out[dst] += alpha * xl[src]; thread handles 4 channels (float4 gather)
__global__ void k_aggregate(const float* __restrict__ ex, const float* __restrict__ denom,
                            const float* __restrict__ xl,
                            const int* __restrict__ src, const int* __restrict__ dst,
                            float* __restrict__ out) {
  int i = blockIdx.x * blockDim.x + threadIdx.x;  // E*64 threads
  if (i >= NEDGES * 64) return;
  int c4 = (i & 15) * 4;
  int h  = (i >> 4) & 3;
  int e  = i >> 6;
  int s = src[e], d = dst[e];
  float alpha = ex[(size_t)e * NH + h] / (denom[(size_t)d * NH + h] + 1e-16f);
  const float* pl = xl  + (size_t)s * HC + h * NC + c4;
  float*       po = out + (size_t)d * HC + h * NC + c4;
  v4f v = *(const v4f*)pl;
  atomicAdd(po + 0, alpha * v[0]);
  atomicAdd(po + 1, alpha * v[1]);
  atomicAdd(po + 2, alpha * v[2]);
  atomicAdd(po + 3, alpha * v[3]);
}

// h = relu(h + b) in place, also emit f16 copy (input to next layer GEMM)
__global__ void k_epilogue(float* __restrict__ h, const float* __restrict__ b,
                           _Float16* __restrict__ hf16) {
  int i = blockIdx.x * blockDim.x + threadIdx.x;
  if (i >= NNODES * HC) return;
  int c = i & 255;
  float v = fmaxf(h[i] + b[c], 0.f);
  h[i] = v;
  hf16[i] = (_Float16)v;
}

// mean-pool accumulation
__global__ void k_pool(const float* __restrict__ h, const int* __restrict__ batch,
                       float* __restrict__ pooled, float* __restrict__ cnt) {
  int i = blockIdx.x * blockDim.x + threadIdx.x;
  if (i >= NNODES * HC) return;
  int c = i & 255, n = i >> 8;
  int g = batch[n];
  atomicAdd(&pooled[(size_t)g * HC + c], h[i]);
  if (c == 0) atomicAdd(&cnt[g], 1.f);
}

// per-graph MLP: 256 -> 128 -> LN -> ReLU -> 64 -> ReLU -> 1
__global__ void k_mlp(const float* __restrict__ pooled, const float* __restrict__ cnt,
                      const float* __restrict__ p1w, const float* __restrict__ p1b,
                      const float* __restrict__ lng, const float* __restrict__ lnb,
                      const float* __restrict__ p2w, const float* __restrict__ p2b,
                      const float* __restrict__ hw,  const float* __restrict__ hb,
                      float* __restrict__ out) {
  __shared__ float gfe[HC];
  __shared__ float z[128];
  __shared__ float red[128];
  int g = blockIdx.x, t = threadIdx.x;        // 128 threads
  float inv = 1.f / fmaxf(cnt[g], 1.f);
  gfe[t]       = pooled[(size_t)g * HC + t] * inv;
  gfe[t + 128] = pooled[(size_t)g * HC + t + 128] * inv;
  __syncthreads();

  float acc = p1b[t];
  for (int k = 0; k < HC; ++k) acc += gfe[k] * p1w[k * 128 + t];
  z[t] = acc;
  red[t] = acc; __syncthreads();
  for (int s = 64; s > 0; s >>= 1) { if (t < s) red[t] += red[t + s]; __syncthreads(); }
  float mu = red[0] / 128.f; __syncthreads();
  float d0 = acc - mu;
  red[t] = d0 * d0; __syncthreads();
  for (int s = 64; s > 0; s >>= 1) { if (t < s) red[t] += red[t + s]; __syncthreads(); }
  float var = red[0] / 128.f; __syncthreads();
  float zn = fmaxf(d0 * rsqrtf(var + 1e-5f) * lng[t] + lnb[t], 0.f);
  z[t] = zn; __syncthreads();

  if (t < 64) {
    float a2 = p2b[t];
    for (int k = 0; k < 128; ++k) a2 += z[k] * p2w[k * 64 + t];
    red[t] = fmaxf(a2, 0.f) * hw[t];
  }
  __syncthreads();
  for (int s = 32; s > 0; s >>= 1) { if (t < s) red[t] += red[t + s]; __syncthreads(); }
  if (t == 0) out[g] = red[0] + hb[0];
}

// ---------------------------------------------------------------------------
// host side
// ---------------------------------------------------------------------------
static inline int gridFor(long n, int b) { return (int)((n + b - 1) / b); }

extern "C" void kernel_launch(void* const* d_in, const int* in_sizes, int n_in,
                              void* d_out, int out_size, void* d_ws, size_t ws_size,
                              hipStream_t stream) {
  const float* x      = (const float*)d_in[0];
  const int*   ei     = (const int*)  d_in[1];
  const float* ea     = (const float*)d_in[2];
  const int*   batch  = (const int*)  d_in[3];
  const float* enc_w  = (const float*)d_in[4];
  const float* enc_b  = (const float*)d_in[5];
  const float* g_wl[2]  = { (const float*)d_in[6],  (const float*)d_in[12] };
  const float* g_bl[2]  = { (const float*)d_in[7],  (const float*)d_in[13] };
  const float* g_wr[2]  = { (const float*)d_in[8],  (const float*)d_in[14] };
  const float* g_we[2]  = { (const float*)d_in[9],  (const float*)d_in[15] };
  const float* g_att[2] = { (const float*)d_in[10], (const float*)d_in[16] };
  const float* g_b[2]   = { (const float*)d_in[11], (const float*)d_in[17] };
  const float* p1w  = (const float*)d_in[18];
  const float* p1b  = (const float*)d_in[19];
  const float* lng  = (const float*)d_in[20];
  const float* lnb  = (const float*)d_in[21];
  const float* p2w  = (const float*)d_in[22];
  const float* p2b  = (const float*)d_in[23];
  const float* hw   = (const float*)d_in[24];
  const float* hb   = (const float*)d_in[25];

  const int* src = ei;
  const int* dst = ei + NEDGES;

  // workspace carve-up (256B aligned)
  char* ws = (char*)d_ws;
  size_t off = 0;
  auto take = [&](size_t bytes) { char* p = ws + off; off = (off + bytes + 255) & ~(size_t)255; return p; };
  _Float16* hf16 = (_Float16*)take((size_t)NNODES * HC * 2);
  float*    xl   = (float*)   take((size_t)NNODES * HC * 4);
  float*    xr   = (float*)   take((size_t)NNODES * HC * 4);
  float*    hout = (float*)   take((size_t)NNODES * HC * 4);
  _Float16* wlt  = (_Float16*)take((size_t)HC * HC * 2);
  _Float16* wrt  = (_Float16*)take((size_t)HC * HC * 2);
  float*    lg   = (float*)   take((size_t)NEDGES * NH * 4);
  unsigned* mxu  = (unsigned*)take((size_t)NNODES * NH * 4);
  float*    den  = (float*)   take((size_t)NNODES * NH * 4);
  float*    pool = (float*)   take((size_t)NG * HC * 4);
  float*    cnt  = (float*)   take((size_t)NG * 4);
  (void)ws_size; (void)n_in; (void)in_sizes; (void)out_size;

  const int B = 256;

  // encoder -> hf16 [N,64]
  k_encoder<<<gridFor((long)NNODES * 64, B), B, 0, stream>>>(x, enc_w, enc_b, hf16);

  int Kl = 64;  // layer-1 GEMM K
  for (int layer = 0; layer < 2; ++layer) {
    // transpose+convert weights to f16 [HC,K]
    k_wconv<<<gridFor((long)Kl * HC, B), B, 0, stream>>>(g_wl[layer], wlt, Kl);
    k_wconv<<<gridFor((long)Kl * HC, B), B, 0, stream>>>(g_wr[layer], wrt, Kl);

    // xl = h @ wl + bl ; xr = h @ wr   (WMMA, 64x64 supertile per wave)
    dim3 gg((NNODES + 63) / 64, HC / 64);
    k_gemm_wmma<<<gg, 32, 0, stream>>>(hf16, wlt, g_bl[layer], xl, Kl);
    k_gemm_wmma<<<gg, 32, 0, stream>>>(hf16, wrt, nullptr,     xr, Kl);

    // reset accumulators
    k_zero_f32<<<gridFor((long)NNODES * HC, B), B, 0, stream>>>(hout, NNODES * HC);
    k_zero_f32<<<gridFor((long)NNODES * NH, B), B, 0, stream>>>((float*)mxu, NNODES * NH);
    k_zero_f32<<<gridFor((long)NNODES * NH, B), B, 0, stream>>>(den, NNODES * NH);

    // attention
    k_edge_logits<<<gridFor((long)NEDGES * NH, B), B, 0, stream>>>(
        xl, xr, ea, g_we[layer], g_att[layer], src, dst, lg);
    k_segmax<<<gridFor((long)NEDGES * NH, B), B, 0, stream>>>(lg, dst, mxu);
    k_expsum<<<gridFor((long)NEDGES * NH, B), B, 0, stream>>>(lg, dst, mxu, den);
    k_aggregate<<<gridFor((long)NEDGES * 64, B), B, 0, stream>>>(lg, den, xl, src, dst, hout);

    // h = relu(out + b); also f16 copy for next GEMM (K becomes 256)
    k_epilogue<<<gridFor((long)NNODES * HC, B), B, 0, stream>>>(hout, g_b[layer], hf16);
    Kl = HC;
  }

  // global mean pool + head MLP
  k_zero_f32<<<gridFor((long)NG * HC, B), B, 0, stream>>>(pool, NG * HC);
  k_zero_f32<<<1, 64, 0, stream>>>(cnt, NG);
  k_pool<<<gridFor((long)NNODES * HC, B), B, 0, stream>>>(hout, batch, pool, cnt);
  k_mlp<<<NG, 128, 0, stream>>>(pool, cnt, p1w, p1b, lng, lnb, p2w, p2b, hw, hb,
                                (float*)d_out);
}
